// KANLayer_29661044146291
// MI455X (gfx1250) — matrix operand order repaired and checked by reading, not compiled, not run on hardware
//
#include <hip/hip_runtime.h>

typedef __attribute__((ext_vector_type(16))) _Float16 v16h;
typedef __attribute__((ext_vector_type(8)))  float    v8f;
typedef __attribute__((ext_vector_type(2)))  _Float16 h2v;
typedef __attribute__((ext_vector_type(4)))  _Float16 h4v;

#define B_SZ 1024
#define I_SZ 64
#define O_SZ 64
#define H_SZ 32
#define BT   64            // batch rows per block
#define THREADS 128        // 4 waves (wave32), one 16-row M-tile each
#define XPAD (BT + 1)      // +1 pad -> conflict-free transposed x tile

__device__ __forceinline__ float leakyf(float v) { return fmaxf(v, 0.01f * v); }

// ---------------------------------------------------------------------------
// Prep kernel: W2 -> f16, pre-swizzled into this kernel's contiguous-k WMMA B-operand lane
// layout: halves[e][t][lane][j] = (f16) W2[e, 16t+(lane&15), 16*(lane>>4)+j].
// Main kernel then loads each B tile with 2 fully-coalesced b128 per lane.
// ---------------------------------------------------------------------------
__global__ __launch_bounds__(256)
void pack_w2_f16(const float* __restrict__ W2, _Float16* __restrict__ dst)
{
    const int e    = blockIdx.x;
    const int n    = threadIdx.x * 4;          // 256 thr * 4 halves = 1024 = H*H
    const int t    = n >> 9;
    const int lane = (n >> 4) & 31;
    const int j    = n & 15;                   // 0,4,8,12
    const int col  = 16 * t + (lane & 15);
    const int hb   = 16 * (lane >> 4);

    const float4 v = *(const float4*)(W2 + (size_t)e * (H_SZ * H_SZ) + col * H_SZ + hb + j);
    h4v h;
    h[0] = (_Float16)v.x; h[1] = (_Float16)v.y;
    h[2] = (_Float16)v.z; h[3] = (_Float16)v.w;
    *(h4v*)(dst + (size_t)e * (H_SZ * H_SZ) + n) = h;
}

// ---------------------------------------------------------------------------
// Helpers
// ---------------------------------------------------------------------------
// sP1 layout: per i, 8 blocks of {W1[4k], b1[4k]} float4 pairs (64 floats/i).
// Lane (hi) needs k in [16hi,16hi+8) and [16hi+8,16hi+16): 4 x ds_load_b128
// per half, broadcast across each 16-lane group; adjacent-VGPR float4 pairs
// let the compiler form v_pk_fma_f32.
__device__ __forceinline__ v16h build_A(const float* __restrict__ sP1,
                                        int i, int hi, float xv)
{
    const h2v c01 = {(_Float16)0.01f, (_Float16)0.01f};
    v16h A;
    const float4* pL = (const float4*)&sP1[(i * 8 + 4 * hi) * 8];      // k = 16hi +0..7
    const float4* pH = (const float4*)&sP1[(i * 8 + 4 * hi + 2) * 8];  // k = 16hi+8+0..7
#pragma unroll
    for (int blk = 0; blk < 2; ++blk) {
        const float4 wL = pL[2 * blk], bL = pL[2 * blk + 1];
        const float4 wH = pH[2 * blk], bH = pH[2 * blk + 1];
        const float y0 = fmaf(xv, wL.x, bL.x), y1 = fmaf(xv, wL.y, bL.y);
        const float y2 = fmaf(xv, wL.z, bL.z), y3 = fmaf(xv, wL.w, bL.w);
        const float z0 = fmaf(xv, wH.x, bH.x), z1 = fmaf(xv, wH.y, bH.y);
        const float z2 = fmaf(xv, wH.z, bH.z), z3 = fmaf(xv, wH.w, bH.w);
        h2v a01; a01[0] = (_Float16)y0; a01[1] = (_Float16)y1;
        h2v a23; a23[0] = (_Float16)y2; a23[1] = (_Float16)y3;
        h2v b01; b01[0] = (_Float16)z0; b01[1] = (_Float16)z1;
        h2v b23; b23[0] = (_Float16)z2; b23[1] = (_Float16)z3;
        a01 = __builtin_elementwise_max(a01, a01 * c01);   // packed leaky
        a23 = __builtin_elementwise_max(a23, a23 * c01);
        b01 = __builtin_elementwise_max(b01, b01 * c01);
        b23 = __builtin_elementwise_max(b23, b23 * c01);
        A[4 * blk + 0] = a01[0]; A[4 * blk + 1] = a01[1];
        A[4 * blk + 2] = a23[0]; A[4 * blk + 3] = a23[1];
        A[8 + 4 * blk + 0] = b01[0]; A[8 + 4 * blk + 1] = b01[1];
        A[8 + 4 * blk + 2] = b23[0]; A[8 + 4 * blk + 3] = b23[1];
    }
    return A;
}

__device__ __forceinline__ void load_B_packed(const _Float16* __restrict__ w2pk,
                                              int e, int lane, v16h& B0, v16h& B1)
{
    const float4* pb = (const float4*)(w2pk + (size_t)e * (H_SZ * H_SZ));
    union { float4 f[2]; v16h v; } u0, u1;
    u0.f[0] = pb[2 * lane];       u0.f[1] = pb[2 * lane + 1];
    u1.f[0] = pb[64 + 2 * lane];  u1.f[1] = pb[64 + 2 * lane + 1];
    B0 = u0.v; B1 = u1.v;
}

__device__ __forceinline__ void load_B_f32(const float* __restrict__ W2,
                                           int e, int l16, int hi, v16h& B0, v16h& B1)
{
    const float* w2b = W2 + (size_t)e * (H_SZ * H_SZ);
    const float4* p0 = (const float4*)(w2b + l16 * H_SZ + 16 * hi);
    const float4* p1 = (const float4*)(w2b + (16 + l16) * H_SZ + 16 * hi);
#pragma unroll
    for (int q = 0; q < 4; ++q) {
        const float4 a = p0[q];
        const float4 b = p1[q];
        B0[4 * q + 0] = (_Float16)a.x;  B0[4 * q + 1] = (_Float16)a.y;
        B0[4 * q + 2] = (_Float16)a.z;  B0[4 * q + 3] = (_Float16)a.w;
        B1[4 * q + 0] = (_Float16)b.x;  B1[4 * q + 1] = (_Float16)b.y;
        B1[4 * q + 2] = (_Float16)b.z;  B1[4 * q + 3] = (_Float16)b.w;
    }
}

// ---------------------------------------------------------------------------
// Main kernel (software-pipelined: WMMA(i+1) issued after epilogue(i), its result
// consumed next iteration, so the next A-build fills the WMMA->VALU hazard window)
// ---------------------------------------------------------------------------
__global__ __launch_bounds__(THREADS)
void kan_layer_wmma_kernel(const float* __restrict__ x,
                           const float* __restrict__ W1,
                           const float* __restrict__ b1,
                           const float* __restrict__ W2,
                           const float* __restrict__ b2,
                           const float* __restrict__ W3,
                           const float* __restrict__ b3,
                           const float* __restrict__ lw,
                           const float* __restrict__ bw,
                           const _Float16* __restrict__ w2pk,
                           const int usePacked,
                           float* __restrict__ out)
{
    __shared__ float sP1[I_SZ * H_SZ * 2];   // {W1,b1} float4-interleaved  16 KB
    __shared__ float sP2[I_SZ * H_SZ * 2];   // {b2,W3} pair-interleaved    16 KB
    __shared__ float sXT[I_SZ * XPAD];       // x transposed, padded        16.25 KB
    __shared__ float sBias[BT];

    const int o     = blockIdx.x & (O_SZ - 1);
    const int bt    = blockIdx.x >> 6;
    const int bBase = bt * BT;

    const int tid  = threadIdx.x;
    const int wave = tid >> 5;
    const int lane = tid & 31;
    const int l16  = lane & 15;
    const int hi   = lane >> 4;

    const int e0     = o * I_SZ;
    const int base32 = e0 * H_SZ;

    // ---- one-time staging ----
    for (int n = tid; n < I_SZ * H_SZ; n += THREADS) {
        const int f = ((n >> 2) << 3) + (n & 3);          // float4-interleave
        sP1[f]     = W1[base32 + n];
        sP1[f + 4] = b1[base32 + n];
        *(float2*)&sP2[2 * n] = make_float2(b2[base32 + n], W3[base32 + n]);
    }
    {
        const int ii = tid & 63;
        for (int r = tid >> 6; r < BT; r += (THREADS >> 6)) {
            sXT[ii * XPAD + r] = x[(size_t)(bBase + r) * I_SZ + ii];
        }
    }
    __syncthreads();

    const int rowL = wave * 16 + l16;

    float accP0[8], accP1[8];
#pragma unroll
    for (int j = 0; j < 8; ++j) { accP0[j] = 0.0f; accP1[j] = 0.0f; }
    float accBias = 0.0f;
    float cB3     = 0.0f;

    // ---- pipeline prologue: stage 0 ----
    float xCur = sXT[rowL];
    v16h A = build_A(sP1, 0, hi, xCur);
    v16h B0, B1;
    if (usePacked) load_B_packed(w2pk, e0, lane, B0, B1);
    else           load_B_f32(W2, e0, l16, hi, B0, B1);
    v8f c0 = {};
    v8f c1 = {};
    c0 = __builtin_amdgcn_wmma_f32_16x16x32_f16(false, A, false, B0, (short)0, c0, false, false);
    c1 = __builtin_amdgcn_wmma_f32_16x16x32_f16(false, A, false, B1, (short)0, c1, false, false);

    for (int i = 0; i < I_SZ; ++i) {
        const int e = e0 + i;

        // ---- build next stage first (independent of c0/c1) ----
        float xNext = 0.0f;
        v16h An, B0n, B1n;
        if (i + 1 < I_SZ) {
            if (i + 2 < I_SZ) {
                const size_t off = (size_t)(e + 2) * (H_SZ * H_SZ) + tid * 8;
                if (usePacked) __builtin_prefetch(w2pk + off, 0, 0);
                else           __builtin_prefetch(W2 + off, 0, 0);
            }
            if (usePacked) load_B_packed(w2pk, e + 1, lane, B0n, B1n);
            else           load_B_f32(W2, e + 1, l16, hi, B0n, B1n);
            xNext = sXT[(i + 1) * XPAD + rowL];
            An = build_A(sP1, i + 1, hi, xNext);
        }

        // ---- epilogue(i): accumulate leaky(c+b2)*(W3*lw) per lane ----
        const float2 q0 = *(const float2*)&sP2[(i * H_SZ + l16) * 2];       // {b2, W3}
        const float2 q1 = *(const float2*)&sP2[(i * H_SZ + 16 + l16) * 2];
        const float lwv = lw[e], b3v = b3[e], bwv = bw[e];
        const float wl0 = q0.y * lwv;
        const float wl1 = q1.y * lwv;
#pragma unroll
        for (int j = 0; j < 8; ++j) {
            const float y0 = leakyf(c0[j] + q0.x);
            const float y1 = leakyf(c1[j] + q1.x);
            accP0[j] = fmaf(y0, wl0, accP0[j]);
            accP1[j] = fmaf(y1, wl1, accP1[j]);
        }
        cB3     = fmaf(lwv, b3v, cB3);
        accBias = fmaf(bwv, leakyf(xCur), accBias);

        // ---- issue next WMMAs (consumed next iteration) ----
        if (i + 1 < I_SZ) {
            v8f z0 = {};
            v8f z1 = {};
            z0 = __builtin_amdgcn_wmma_f32_16x16x32_f16(false, An, false, B0n, (short)0, z0, false, false);
            z1 = __builtin_amdgcn_wmma_f32_16x16x32_f16(false, An, false, B1n, (short)0, z1, false, false);
            c0 = z0; c1 = z1;
        }
        xCur = xNext;
    }

    // ---- one-time cross-lane row reduction (within each 16-lane half) ----
    float red[8];
#pragma unroll
    for (int j = 0; j < 8; ++j) {
        float s = accP0[j] + accP1[j];
        s += __shfl_xor(s, 1, 32);
        s += __shfl_xor(s, 2, 32);
        s += __shfl_xor(s, 4, 32);
        s += __shfl_xor(s, 8, 32);
        red[j] = s;
    }

    if (lane < 16) sBias[wave * 16 + lane] = accBias;
    __syncthreads();

    if (l16 == 0) {
#pragma unroll
        for (int j = 0; j < 8; ++j) {
            const int r = wave * 16 + 8 * hi + j;
            out[(size_t)(bBase + r) * O_SZ + o] = red[j] + cB3 + sBias[r];
        }
    }
}

extern "C" void kernel_launch(void* const* d_in, const int* in_sizes, int n_in,
                              void* d_out, int out_size, void* d_ws, size_t ws_size,
                              hipStream_t stream) {
    (void)in_sizes; (void)n_in; (void)out_size;

    const float* x  = (const float*)d_in[0];
    const float* W1 = (const float*)d_in[1];
    const float* b1 = (const float*)d_in[2];
    const float* W2 = (const float*)d_in[3];
    const float* b2 = (const float*)d_in[4];
    const float* W3 = (const float*)d_in[5];
    const float* b3 = (const float*)d_in[6];
    const float* lw = (const float*)d_in[7];
    const float* bw = (const float*)d_in[8];
    float* out = (float*)d_out;

    const size_t packBytes = (size_t)O_SZ * I_SZ * H_SZ * H_SZ * 2u;   // 8 MB f16
    const int usePacked = (d_ws != nullptr && ws_size >= packBytes) ? 1 : 0;
    _Float16* w2pk = (_Float16*)d_ws;

    if (usePacked) {
        pack_w2_f16<<<O_SZ * I_SZ, 256, 0, stream>>>(W2, w2pk);
    }

    const int grid = O_SZ * (B_SZ / BT);   // 1024 blocks
    kan_layer_wmma_kernel<<<grid, THREADS, 0, stream>>>(
        x, W1, b1, W2, b2, W3, b3, lw, bw, w2pk, usePacked, out);
}
